// OCAB_54975581388991
// MI455X (gfx1250) — compile-verified, matrix operand order checked by reading, not run on hardware
//
#include <hip/hip_runtime.h>
#include <hip/hip_bf16.h>
#include <math.h>

// ---------------------------------------------------------------------------
// OCAB block for MI455X (gfx1250, wave32, WMMA f32_16x16x32_f16)
// L=65536 tokens, C=180 (6 heads x 30), WS=16, OWS=24, MLP_H=360
// GEMMs: double-buffered async LDS DMA (global_load_async_to_lds_b128,
// ASYNCcnt) overlapping DMA of step i+1 with WMMAs of step i.
// ---------------------------------------------------------------------------

typedef __attribute__((ext_vector_type(16))) _Float16 v16h;
typedef __attribute__((ext_vector_type(8)))  _Float16 v8h;
typedef __attribute__((ext_vector_type(8)))  float    v8f;

#define WMMA_F16(a, b, c) \
  __builtin_amdgcn_wmma_f32_16x16x32_f16(false, (a), false, (b), (short)0, (c), false, false)

#define L_TOK   65536
#define CDIM    180
#define CPAD    192
#define NHEAD   6
#define HDIM    30
#define HPAD    32
#define NKEY    576
#define QSCALE  0.18257418583505536f   // 30^-0.5

#define CAT16(lo, hi) \
  __builtin_shufflevector((lo), (hi), 0, 1, 2, 3, 4, 5, 6, 7, 8, 9, 10, 11, 12, 13, 14, 15)

// ---- gfx1250 async LDS DMA (ASYNCcnt path, CDNA5 ISA) ---
// One 16B global->LDS transfer per lane, no VGPR transit.
__device__ __forceinline__ void async_copy16(unsigned lds, const _Float16* g) {
  asm volatile("global_load_async_to_lds_b128 %0, %1, off"
               :: "v"(lds), "v"((unsigned long long)(uintptr_t)g)
               : "memory");
}
#define WAIT_ASYNC0() asm volatile("s_wait_asynccnt 0" ::: "memory")

// ---- WMMA fragment loaders --------------------------------------------------
// A (16 rows x 32 K): lane -> row (lane&15); per lane the halves are two
// contiguous 16B runs at K = kb..kb+7 and kb+16..kb+23 (kb = 8*(lane>=16)).
__device__ __forceinline__ v16h load_A_lds(const _Float16* s, int ld) {
  const int lane = threadIdx.x & 31;
  const int row  = lane & 15;
  const int kb   = (lane & 16) ? 8 : 0;
  const v8h lo = *(const v8h*)(s + row * ld + kb);
  const v8h hi = *(const v8h*)(s + row * ld + kb + 16);
  return CAT16(lo, hi);
}

// B (32 K x 16 cols) stored column-major in LDS (stride 32 halves per column).
__device__ __forceinline__ v16h load_B_ldsT(const _Float16* s) {
  const int lane = threadIdx.x & 31;
  const int col  = lane & 15;
  const int kb   = (lane & 16) ? 8 : 0;
  const v8h lo = *(const v8h*)(s + col * 32 + kb);
  const v8h hi = *(const v8h*)(s + col * 32 + kb + 16);
  return CAT16(lo, hi);
}

// ---------------------------------------------------------------------------
// K0: pack weights -> transposed, zero-padded f16: out[c*Kpad + k]
// headRemap: K rows are per-head padded (k -> head=k/32, dd=k%32, pad dd>=30)
// ---------------------------------------------------------------------------
__global__ __launch_bounds__(256) void pack_w(
    const float* __restrict__ in, _Float16* __restrict__ out,
    int K, int C, int Kpad, int Npad, int headRemap) {
  const int idx = blockIdx.x * 256 + threadIdx.x;
  if (idx >= Npad * Kpad) return;
  const int c = idx / Kpad, k = idx % Kpad;
  int srcRow, valid;
  if (headRemap) { int hh = k >> 5, dd = k & 31; srcRow = hh * HDIM + dd; valid = (dd < HDIM); }
  else           { srcRow = k; valid = (k < K); }
  out[idx] = (_Float16)((valid && c < C) ? in[srcRow * C + c] : 0.f);
}

// ---------------------------------------------------------------------------
// K1/K6: LayerNorm (f32 in) -> f16 out padded to CPAD with zeros
// ---------------------------------------------------------------------------
__global__ __launch_bounds__(256) void ln_to_f16(
    const float* __restrict__ x, const float* __restrict__ w,
    const float* __restrict__ b, _Float16* __restrict__ out) {
  const int t = blockIdx.x * blockDim.x + threadIdx.x;   // token
  const float* xr = x + (size_t)t * CDIM;
  const float4* x4 = (const float4*)xr;                  // 720B row, 16B aligned
  __builtin_prefetch(xr + CDIM, 0, 1);                   // next token's row
  float s = 0.f, ss = 0.f;
  for (int i = 0; i < CDIM / 4; ++i) {
    float4 v = x4[i];
    s  += v.x + v.y + v.z + v.w;
    ss += v.x * v.x + v.y * v.y + v.z * v.z + v.w * v.w;
  }
  const float m   = s * (1.0f / CDIM);
  const float var = ss * (1.0f / CDIM) - m * m;
  const float inv = rsqrtf(var + 1e-5f);
  _Float16* o = out + (size_t)t * CPAD;
  for (int i = 0; i < CDIM; ++i) o[i] = (_Float16)((xr[i] - m) * inv * w[i] + b[i]);
  for (int i = CDIM; i < CPAD; ++i) o[i] = (_Float16)0.f;
}

// ---------------------------------------------------------------------------
// K3: expand relative-position bias: bias[h][q][n] = rpb[rpi[q][n]*6 + h]
// ---------------------------------------------------------------------------
__global__ __launch_bounds__(256) void expand_bias(
    const int* __restrict__ rpi, const float* __restrict__ rpb,
    float* __restrict__ bias) {
  const int idx = blockIdx.x * 256 + threadIdx.x;        // 6*256*576 total
  const int n   = idx % NKEY;
  const int rem = idx / NKEY;
  const int q   = rem % 256;
  const int h   = rem / 256;
  bias[idx] = rpb[rpi[q * NKEY + n] * NHEAD + h];
}

// ---------------------------------------------------------------------------
// K2: QKV GEMM. xn_f16 (L x 192) @ Wqkv_t (576 x 192, padded) -> scatter into
// per-head padded f16 layouts Qp/Kp/Vp (L x 6 x 32), Q pre-scaled.
// ---------------------------------------------------------------------------
__device__ __forceinline__ void qkv_store(
    _Float16* Qp, _Float16* Kp, _Float16* Vp, const float* qb,
    int token, int c, float v) {
  if (c >= 3 * CDIM) return;
  v += qb[c];
  const int which = c / CDIM;
  const int cc    = c % CDIM;
  const int h     = cc / HDIM;
  const int dd    = cc % HDIM;
  const size_t off = (size_t)token * CPAD + h * HPAD + dd;
  if      (which == 0) Qp[off] = (_Float16)(v * QSCALE);
  else if (which == 1) Kp[off] = (_Float16)v;
  else                 Vp[off] = (_Float16)v;
}

__global__ __launch_bounds__(256) void gemm_qkv(
    const _Float16* __restrict__ A, const _Float16* __restrict__ Wt,
    const float* __restrict__ qb,
    _Float16* __restrict__ Qp, _Float16* __restrict__ Kp,
    _Float16* __restrict__ Vp) {
  __shared__ __align__(16) _Float16 As[2][64 * 32];
  __shared__ __align__(16) _Float16 Bs[2][64 * 32];
  const int t = threadIdx.x;
  const int rowBase = blockIdx.x * 64;
  const int colBase = blockIdx.y * 64;
  const int wv = t >> 5, lane = t & 31;
  const int wm = wv >> 1, wn = wv & 1;
  const int ar = t >> 2, ak = (t & 3) * 8;         // 16B per thread
  const unsigned asb = (unsigned)(uintptr_t)(&As[0][0]) + t * 16;
  const unsigned bsb = (unsigned)(uintptr_t)(&Bs[0][0]) + t * 16;
  const _Float16* aSrc = A  + (size_t)(rowBase + ar) * CPAD + ak;
  const _Float16* bSrc = Wt + (size_t)(colBase + ar) * CPAD + ak;
  v8f acc0 = {}, acc1 = {};
  async_copy16(asb, aSrc);                          // stage step 0
  async_copy16(bsb, bSrc);
  const int NSTEP = CPAD / 32;
  for (int i = 0; i < NSTEP; ++i) {
    WAIT_ASYNC0();
    __syncthreads();
    if (i + 1 < NSTEP) {                            // DMA step i+1 under compute
      const unsigned nb = ((i + 1) & 1) * (64 * 32 * 2);
      async_copy16(asb + nb, aSrc + (i + 1) * 32);
      async_copy16(bsb + nb, bSrc + (i + 1) * 32);
    }
    const _Float16* as = &As[i & 1][0];
    const _Float16* bs = &Bs[i & 1][0];
    v16h a  = load_A_lds(as + wm * 16 * 32, 32);
    v16h b0 = load_B_ldsT(bs + (wn * 32) * 32);
    v16h b1 = load_B_ldsT(bs + (wn * 32 + 16) * 32);
    acc0 = WMMA_F16(a, b0, acc0);
    acc1 = WMMA_F16(a, b1, acc1);
  }
  const int hi = (lane & 16) ? 8 : 0;
  const int c0 = colBase + wn * 32 + (lane & 15);
#pragma unroll
  for (int r = 0; r < 8; ++r) {
    int token = rowBase + wm * 16 + r + hi;
    qkv_store(Qp, Kp, Vp, qb, token, c0,      acc0[r]);
    qkv_store(Qp, Kp, Vp, qb, token, c0 + 16, acc1[r]);
  }
}

// ---------------------------------------------------------------------------
// K4: overlapping-window flash attention. Block = (window, head), 8 waves x
// 2 q-tiles. Key chunks of 32 staged in LDS (K row-major, V transposed) so
// every WMMA operand is a pair of ds_load_b128.
// ---------------------------------------------------------------------------
__global__ __launch_bounds__(256) void attn_kernel(
    const _Float16* __restrict__ Qp, const _Float16* __restrict__ Kp,
    const _Float16* __restrict__ Vp, const float* __restrict__ bias,
    _Float16* __restrict__ attn_out) {
  __shared__ __align__(16) _Float16 Ks[32 * 32];          // [key][dim]
  __shared__ __align__(16) _Float16 Vs[32 * 32];          // [dim][key]
  __shared__ __align__(16) _Float16 pbuf[8][16 * 32];
  const int win  = blockIdx.x;
  const int head = blockIdx.y;
  const int wy = win >> 4, wx = win & 15;
  const int t = threadIdx.x, wv = t >> 5, lane = t & 31;
  const int laneHi = (lane & 16) ? 1 : 0;
  const int col = lane & 15;
  const int kb8 = laneHi ? 8 : 0;
  const float* biasH = bias + (size_t)head * 256 * NKEY;
  _Float16* pb = &pbuf[wv][0];

  // per-wave flash state for both q-tiles
  v16h qa[2];
  v8f o0[2], o1[2];
  float mrun[2][8], lrun[2][8];
#pragma unroll
  for (int s = 0; s < 2; ++s) {
    const int qt = wv * 2 + s;
    const int qq = qt * 16 + (lane & 15);
    const int ty = wy * 16 + (qq >> 4), tx = wx * 16 + (qq & 15);
    const size_t qoff = (size_t)(ty * 256 + tx) * CPAD + head * HPAD;
    qa[s] = CAT16(*(const v8h*)(Qp + qoff + kb8), *(const v8h*)(Qp + qoff + kb8 + 16));
    v8f z = {};
    o0[s] = z; o1[s] = z;
#pragma unroll
    for (int r = 0; r < 8; ++r) { mrun[s][r] = -1e30f; lrun[s][r] = 0.f; }
  }

  for (int kc = 0; kc < NKEY / 32; ++kc) {
    const int nb = kc * 32;
    __syncthreads();
    // ---- cooperative staging: 32 keys x 32 dims each for K and V ----------
    {
      const int u   = t & 127;
      const int key = u >> 2, part = (u & 3) * 8;
      const int nn  = nb + key;
      const int ky  = wy * 16 - 4 + nn / 24;
      const int kx  = wx * 16 - 4 + nn % 24;
      const bool ok = (ky >= 0 && ky < 256 && kx >= 0 && kx < 256);
      const size_t base = (size_t)(ky * 256 + kx) * CPAD + head * HPAD + part;
      if (t < 128) {                       // K: vector copy, [key][dim]
        v8h v = {};
        if (ok) v = *(const v8h*)(Kp + base);
        *(v8h*)&Ks[key * 32 + part] = v;
      } else {                             // V: vector load, transposed scatter
        v8h v = {};
        if (ok) v = *(const v8h*)(Vp + base);
#pragma unroll
        for (int j = 0; j < 8; ++j) Vs[(part + j) * 32 + key] = v[j];
      }
    }
    __syncthreads();
    // ---- B fragments straight from LDS (2x ds_load_b128 each) -------------
    const v16h bk0 = load_B_ldsT(&Ks[0]);           // keys 0..15
    const v16h bk1 = load_B_ldsT(&Ks[16 * 32]);     // keys 16..31
    const v16h bv0 = load_B_ldsT(&Vs[0]);           // dims 0..15
    const v16h bv1 = load_B_ldsT(&Vs[16 * 32]);     // dims 16..31
    const int n0 = nb + col;

#pragma unroll
    for (int s = 0; s < 2; ++s) {
      const int qt = wv * 2 + s;
      v8f z = {};
      v8f s0 = WMMA_F16(qa[s], bk0, z);
      v8f s1 = WMMA_F16(qa[s], bk1, z);
      float p0[8], p1[8];
#pragma unroll
      for (int r = 0; r < 8; ++r) {
        const int q = qt * 16 + r + (laneHi ? 8 : 0);
        const float* bq = biasH + (size_t)q * NKEY;
        s0[r] += bq[n0];
        s1[r] += bq[n0 + 16];
        float cm = fmaxf(s0[r], s1[r]);
#pragma unroll
        for (int off = 1; off < 16; off <<= 1) cm = fmaxf(cm, __shfl_xor(cm, off, 32));
        const float mn    = fmaxf(mrun[s][r], cm);
        const float alpha = __expf(mrun[s][r] - mn);
        const float e0 = __expf(s0[r] - mn);
        const float e1 = __expf(s1[r] - mn);
        float rs = e0 + e1;
#pragma unroll
        for (int off = 1; off < 16; off <<= 1) rs += __shfl_xor(rs, off, 32);
        lrun[s][r] = lrun[s][r] * alpha + rs;
        mrun[s][r] = mn;
        o0[s][r] *= alpha; o1[s][r] *= alpha;
        p0[r] = e0; p1[r] = e1;
      }
      // P: D-layout -> LDS -> A-layout (cross-lane transpose)
#pragma unroll
      for (int r = 0; r < 8; ++r) {
        const int row = r + (laneHi ? 8 : 0);
        pb[row * 32 + col]      = (_Float16)p0[r];
        pb[row * 32 + col + 16] = (_Float16)p1[r];
      }
      v16h pa = load_A_lds(pb, 32);
      o0[s] = WMMA_F16(pa, bv0, o0[s]);
      o1[s] = WMMA_F16(pa, bv1, o1[s]);
    }
  }
  // finalize: /l and store to padded per-head f16 layout
#pragma unroll
  for (int s = 0; s < 2; ++s) {
    const int qt = wv * 2 + s;
#pragma unroll
    for (int r = 0; r < 8; ++r) {
      const float inv = 1.f / lrun[s][r];
      const int qq = qt * 16 + r + (laneHi ? 8 : 0);
      const int ty = wy * 16 + (qq >> 4), tx = wx * 16 + (qq & 15);
      const size_t ob = (size_t)(ty * 256 + tx) * CPAD + head * HPAD;
      attn_out[ob + col]      = (_Float16)(o0[s][r] * inv);
      attn_out[ob + col + 16] = (_Float16)((col + 16 < HDIM) ? o1[s][r] * inv : 0.f);
    }
  }
}

// ---------------------------------------------------------------------------
// K5: proj GEMM + bias + residual -> x2 (f32). Wt pre-packed 192x192.
// ---------------------------------------------------------------------------
__global__ __launch_bounds__(256) void gemm_proj(
    const _Float16* __restrict__ A, const _Float16* __restrict__ Wt,
    const float* __restrict__ pb, const float* __restrict__ xres,
    float* __restrict__ x2) {
  __shared__ __align__(16) _Float16 As[2][64 * 32];
  __shared__ __align__(16) _Float16 Bs[2][64 * 32];
  const int t = threadIdx.x;
  const int rowBase = blockIdx.x * 64;
  const int colBase = blockIdx.y * 64;
  const int wv = t >> 5, lane = t & 31;
  const int wm = wv >> 1, wn = wv & 1;
  const int ar = t >> 2, ak = (t & 3) * 8;
  const unsigned asb = (unsigned)(uintptr_t)(&As[0][0]) + t * 16;
  const unsigned bsb = (unsigned)(uintptr_t)(&Bs[0][0]) + t * 16;
  const _Float16* aSrc = A  + (size_t)(rowBase + ar) * CPAD + ak;
  const _Float16* bSrc = Wt + (size_t)(colBase + ar) * CPAD + ak;
  v8f acc0 = {}, acc1 = {};
  async_copy16(asb, aSrc);
  async_copy16(bsb, bSrc);
  const int NSTEP = CPAD / 32;
  for (int i = 0; i < NSTEP; ++i) {
    WAIT_ASYNC0();
    __syncthreads();
    if (i + 1 < NSTEP) {
      const unsigned nb = ((i + 1) & 1) * (64 * 32 * 2);
      async_copy16(asb + nb, aSrc + (i + 1) * 32);
      async_copy16(bsb + nb, bSrc + (i + 1) * 32);
    }
    const _Float16* as = &As[i & 1][0];
    const _Float16* bs = &Bs[i & 1][0];
    v16h a  = load_A_lds(as + wm * 16 * 32, 32);
    v16h b0 = load_B_ldsT(bs + (wn * 32) * 32);
    v16h b1 = load_B_ldsT(bs + (wn * 32 + 16) * 32);
    acc0 = WMMA_F16(a, b0, acc0);
    acc1 = WMMA_F16(a, b1, acc1);
  }
  const int hi = (lane & 16) ? 8 : 0;
  const int c0 = colBase + wn * 32 + (lane & 15);
#pragma unroll
  for (int r = 0; r < 8; ++r) {
    const int token = rowBase + wm * 16 + r + hi;
    if (c0 < CDIM)
      x2[(size_t)token * CDIM + c0] = acc0[r] + pb[c0] + xres[(size_t)token * CDIM + c0];
    const int c1 = c0 + 16;
    if (c1 < CDIM)
      x2[(size_t)token * CDIM + c1] = acc1[r] + pb[c1] + xres[(size_t)token * CDIM + c1];
  }
}

// ---------------------------------------------------------------------------
// K7: fc1 GEMM + exact GELU -> h_f16 (L x 384, K-padded). Wt 384x192.
// ---------------------------------------------------------------------------
__device__ __forceinline__ float gelu_exact(float v) {
  return 0.5f * v * (1.0f + erff(v * 0.7071067811865476f));
}

__global__ __launch_bounds__(256) void gemm_fc1(
    const _Float16* __restrict__ A, const _Float16* __restrict__ Wt,
    const float* __restrict__ b1, _Float16* __restrict__ hb) {
  __shared__ __align__(16) _Float16 As[2][64 * 32];
  __shared__ __align__(16) _Float16 Bs[2][64 * 32];
  const int t = threadIdx.x;
  const int rowBase = blockIdx.x * 64;
  const int colBase = blockIdx.y * 64;
  const int wv = t >> 5, lane = t & 31;
  const int wm = wv >> 1, wn = wv & 1;
  const int ar = t >> 2, ak = (t & 3) * 8;
  const unsigned asb = (unsigned)(uintptr_t)(&As[0][0]) + t * 16;
  const unsigned bsb = (unsigned)(uintptr_t)(&Bs[0][0]) + t * 16;
  const _Float16* aSrc = A  + (size_t)(rowBase + ar) * CPAD + ak;
  const _Float16* bSrc = Wt + (size_t)(colBase + ar) * CPAD + ak;
  v8f acc0 = {}, acc1 = {};
  async_copy16(asb, aSrc);
  async_copy16(bsb, bSrc);
  const int NSTEP = CPAD / 32;
  for (int i = 0; i < NSTEP; ++i) {
    WAIT_ASYNC0();
    __syncthreads();
    if (i + 1 < NSTEP) {
      const unsigned nb = ((i + 1) & 1) * (64 * 32 * 2);
      async_copy16(asb + nb, aSrc + (i + 1) * 32);
      async_copy16(bsb + nb, bSrc + (i + 1) * 32);
    }
    const _Float16* as = &As[i & 1][0];
    const _Float16* bs = &Bs[i & 1][0];
    v16h a   = load_A_lds(as + wm * 16 * 32, 32);
    v16h b0  = load_B_ldsT(bs + (wn * 32) * 32);
    v16h b1f = load_B_ldsT(bs + (wn * 32 + 16) * 32);
    acc0 = WMMA_F16(a, b0, acc0);
    acc1 = WMMA_F16(a, b1f, acc1);
  }
  const int hi = (lane & 16) ? 8 : 0;
  const int c0 = colBase + wn * 32 + (lane & 15);
#pragma unroll
  for (int r = 0; r < 8; ++r) {
    const int token = rowBase + wm * 16 + r + hi;
    float v0 = (c0 < 360) ? gelu_exact(acc0[r] + b1[c0]) : 0.f;
    hb[(size_t)token * 384 + c0] = (_Float16)v0;
    const int c1 = c0 + 16;
    float v1 = (c1 < 360) ? gelu_exact(acc1[r] + b1[c1]) : 0.f;
    hb[(size_t)token * 384 + c1] = (_Float16)v1;
  }
}

// ---------------------------------------------------------------------------
// K8: fc2 GEMM + bias + residual -> d_out (f32). Wt 192x384, A ld 384.
// ---------------------------------------------------------------------------
__global__ __launch_bounds__(256) void gemm_fc2(
    const _Float16* __restrict__ A, const _Float16* __restrict__ Wt,
    const float* __restrict__ b2, const float* __restrict__ x2,
    float* __restrict__ out) {
  __shared__ __align__(16) _Float16 As[2][64 * 32];
  __shared__ __align__(16) _Float16 Bs[2][64 * 32];
  const int t = threadIdx.x;
  const int rowBase = blockIdx.x * 64;
  const int colBase = blockIdx.y * 64;
  const int wv = t >> 5, lane = t & 31;
  const int wm = wv >> 1, wn = wv & 1;
  const int ar = t >> 2, ak = (t & 3) * 8;
  const unsigned asb = (unsigned)(uintptr_t)(&As[0][0]) + t * 16;
  const unsigned bsb = (unsigned)(uintptr_t)(&Bs[0][0]) + t * 16;
  const _Float16* aSrc = A  + (size_t)(rowBase + ar) * 384 + ak;
  const _Float16* bSrc = Wt + (size_t)(colBase + ar) * 384 + ak;
  v8f acc0 = {}, acc1 = {};
  async_copy16(asb, aSrc);
  async_copy16(bsb, bSrc);
  const int NSTEP = 384 / 32;
  for (int i = 0; i < NSTEP; ++i) {
    WAIT_ASYNC0();
    __syncthreads();
    if (i + 1 < NSTEP) {
      const unsigned nb = ((i + 1) & 1) * (64 * 32 * 2);
      async_copy16(asb + nb, aSrc + (i + 1) * 32);
      async_copy16(bsb + nb, bSrc + (i + 1) * 32);
    }
    const _Float16* as = &As[i & 1][0];
    const _Float16* bs = &Bs[i & 1][0];
    v16h a  = load_A_lds(as + wm * 16 * 32, 32);
    v16h b0 = load_B_ldsT(bs + (wn * 32) * 32);
    v16h b1 = load_B_ldsT(bs + (wn * 32 + 16) * 32);
    acc0 = WMMA_F16(a, b0, acc0);
    acc1 = WMMA_F16(a, b1, acc1);
  }
  const int hi = (lane & 16) ? 8 : 0;
  const int c0 = colBase + wn * 32 + (lane & 15);
#pragma unroll
  for (int r = 0; r < 8; ++r) {
    const int token = rowBase + wm * 16 + r + hi;
    if (c0 < CDIM)
      out[(size_t)token * CDIM + c0] = acc0[r] + b2[c0] + x2[(size_t)token * CDIM + c0];
    const int c1 = c0 + 16;
    if (c1 < CDIM)
      out[(size_t)token * CDIM + c1] = acc1[r] + b2[c1] + x2[(size_t)token * CDIM + c1];
  }
}

// ---------------------------------------------------------------------------
// Launch. Workspace layout (~155.1 MB):
//   [0]        xn_f16 (L*192 f16)  -> reused as attn_f16 after K2
//   [+SEG]     Qp                  -> reused as xn2_f16 after K4
//   [+2SEG]    Kp  \ reused as x2 (L*180 f32) after attention
//   [+3SEG]    Vp  /
//   [+4SEG]    bias (6*256*576 f32)
//   [+..]      h_f16 (L*384 f16)
//   [+..]      packed weights: Wqkv_t(576x192) Wproj_t(192x192)
//              Wfc1_t(384x192) Wfc2_t(192x384)   (f16, ~590KB)
// ---------------------------------------------------------------------------
extern "C" void kernel_launch(void* const* d_in, const int* in_sizes, int n_in,
                              void* d_out, int out_size, void* d_ws, size_t ws_size,
                              hipStream_t stream) {
  const float* x     = (const float*)d_in[0];
  const int*   rpi   = (const int*)d_in[1];
  const float* n1w   = (const float*)d_in[2];
  const float* n1b   = (const float*)d_in[3];
  const float* qkvw  = (const float*)d_in[4];
  const float* qkvb  = (const float*)d_in[5];
  const float* rpb   = (const float*)d_in[6];
  const float* projw = (const float*)d_in[7];
  const float* projb = (const float*)d_in[8];
  const float* n2w   = (const float*)d_in[9];
  const float* n2b   = (const float*)d_in[10];
  const float* fc1w  = (const float*)d_in[11];
  const float* fc1b  = (const float*)d_in[12];
  const float* fc2w  = (const float*)d_in[13];
  const float* fc2b  = (const float*)d_in[14];
  float* out = (float*)d_out;

  char* ws = (char*)d_ws;
  const size_t SEG = (size_t)L_TOK * CPAD * sizeof(_Float16);   // 25,165,824
  _Float16* xn   = (_Float16*)(ws);
  _Float16* Qp   = (_Float16*)(ws + SEG);
  _Float16* Kp   = (_Float16*)(ws + 2 * SEG);
  _Float16* Vp   = (_Float16*)(ws + 3 * SEG);
  float*    bias = (float*)(ws + 4 * SEG);
  char*     p    = ws + 4 * SEG + (size_t)NHEAD * 256 * NKEY * 4;
  _Float16* hb   = (_Float16*)p;            p += (size_t)L_TOK * 384 * 2;
  _Float16* Wqkv = (_Float16*)p;            p += 576 * 192 * 2;
  _Float16* Wproj= (_Float16*)p;            p += 192 * 192 * 2;
  _Float16* Wfc1 = (_Float16*)p;            p += 384 * 192 * 2;
  _Float16* Wfc2 = (_Float16*)p;            p += 192 * 384 * 2;
  // region reuse (kernels are strictly ordered on the stream):
  _Float16* attn = xn;                         // xn dead after K2
  _Float16* xn2  = Qp;                         // Qp dead after K4
  float*    x2   = (float*)(ws + 2 * SEG);     // Kp+Vp dead after K4 (47.2MB fits)

  // zero Q/K/V so the per-head pad lanes (dd=30,31) read as exact zeros
  hipMemsetAsync(Qp, 0, 3 * SEG, stream);

  // pack weights (transposed + zero-padded f16; branch-free GEMM B loads)
  pack_w<<<(576 * 192) / 256, 256, 0, stream>>>(qkvw,  Wqkv,  CDIM, 540, CPAD, 576, 0);
  pack_w<<<(192 * 192) / 256, 256, 0, stream>>>(projw, Wproj, CDIM, CDIM, CPAD, 192, 1);
  pack_w<<<(384 * 192) / 256, 256, 0, stream>>>(fc1w,  Wfc1,  CDIM, 360, CPAD, 384, 0);
  pack_w<<<(192 * 384) / 256, 256, 0, stream>>>(fc2w,  Wfc2,  360, CDIM, 384, 192, 0);

  ln_to_f16  <<<L_TOK / 256, 256, 0, stream>>>(x, n1w, n1b, xn);
  expand_bias<<<(NHEAD * 256 * NKEY) / 256, 256, 0, stream>>>(rpi, rpb, bias);
  gemm_qkv   <<<dim3(L_TOK / 64, 9), 256, 0, stream>>>(xn, Wqkv, qkvb, Qp, Kp, Vp);
  attn_kernel<<<dim3(256, NHEAD), 256, 0, stream>>>(Qp, Kp, Vp, bias, attn);
  gemm_proj  <<<dim3(L_TOK / 64, 3), 256, 0, stream>>>(attn, Wproj, projb, x, x2);
  ln_to_f16  <<<L_TOK / 256, 256, 0, stream>>>(x2, n2w, n2b, xn2);
  gemm_fc1   <<<dim3(L_TOK / 64, 6), 256, 0, stream>>>(xn2, Wfc1, fc1b, hb);
  gemm_fc2   <<<dim3(L_TOK / 64, 3), 256, 0, stream>>>(hb, Wfc2, fc2b, x2, out);
}
